// TaskAlignedAssigner_43001212567929
// MI455X (gfx1250) — matrix operand order repaired and checked by reading, not compiled, not run on hardware
//
#include <hip/hip_runtime.h>
#include <hip/hip_bf16.h>
#include <stdint.h>

// ---------------- problem constants (from reference) ----------------
#define B_    32
#define A_    8400
#define NB_   32
#define C_    80
#define TOPK_ 10

typedef __attribute__((ext_vector_type(16))) _Float16     v16h;
typedef __attribute__((ext_vector_type(8)))  float        v8f;
typedef __attribute__((ext_vector_type(4)))  unsigned int v4u;
typedef __attribute__((ext_vector_type(8)))  int          v8i;
typedef __attribute__((ext_vector_type(4)))  int          v4i;

#if __has_builtin(__builtin_amdgcn_tensor_load_to_lds)
#define HAVE_TDM 1
#else
#define HAVE_TDM 0
#endif

// =====================================================================
// K0: bcs = onehot(gt_labels) @ pred_scores^T   (per batch: 32x80 @ 80x8400)
// One wave (32 lanes) per 16-anchor tile; two 16x16 M-tiles (g 0-15, 16-31)
// share the B operand. v_wmma_f32_16x16x32_f16, K padded 80 -> 96.
// =====================================================================
__global__ __launch_bounds__(32)
void k0_bcs_wmma(const float* __restrict__ scores,   // (B, A, C)
                 const int*   __restrict__ labels,   // (B, NB)
                 float*       __restrict__ bcs)      // (B, NB, A)
{
    const int b    = blockIdx.y;
    const int a0   = blockIdx.x * 16;
    const int lane = threadIdx.x;        // 0..31
    const int n    = lane & 15;          // anchor column within tile
    const int kh   = lane >> 4;          // 0: K 0..15 half, 1: K 16..31 half
    const int aCol = a0 + n;             // A_ = 525*16 exactly, no tail

    const int cls0 = labels[b * NB_ + n];        // rows 0..15  (M-tile 0)
    const int cls1 = labels[b * NB_ + 16 + n];   // rows 16..31 (M-tile 1)

    v8f c0 = {};
    v8f c1 = {};

    #pragma unroll
    for (int kstep = 0; kstep < 3; ++kstep) {
        // ---- B operand: 32x16 f16 tile of scores^T.
        // Lane layout (contiguous K order used here, same as A): lanes 0-15 hold local K=0..15, lanes
        // 16-31 hold K=16..31; element j <-> local K=j, column N = lane%16.
        const int kbase = kstep * 32 + kh * 16;
        v16h bm;
        if (kbase < C_) {
            // kbase in {0,16,32,48,64}; kbase+15 <= 79 -> fully in-row.
            const float4* sp4 =
                (const float4*)(scores + ((size_t)b * A_ + aCol) * C_ + kbase);
            #pragma unroll
            for (int q = 0; q < 4; ++q) {
                float4 f = sp4[q];
                bm[4*q + 0] = (_Float16)f.x;
                bm[4*q + 1] = (_Float16)f.y;
                bm[4*q + 2] = (_Float16)f.z;
                bm[4*q + 3] = (_Float16)f.w;
            }
        } else {                          // kbase == 80: zero padding (K 80..95)
            #pragma unroll
            for (int j = 0; j < 16; ++j) bm[j] = (_Float16)0.0f;
        }

        // ---- A operands: one-hot rows, 16x32 f16 layout (matches B):
        // lane<16 : element h -> K = h
        // lane>=16: element h -> K = h + 16
        v16h am0, am1;
        #pragma unroll
        for (int h = 0; h < 16; ++h) {
            const int kg = kstep * 32 + h + kh * 16;
            am0[h] = (cls0 == kg) ? (_Float16)1.0f : (_Float16)0.0f;
            am1[h] = (cls1 == kg) ? (_Float16)1.0f : (_Float16)0.0f;
        }

        c0 = __builtin_amdgcn_wmma_f32_16x16x32_f16(false, am0, false, bm,
                                                    (short)0, c0, false, false);
        c1 = __builtin_amdgcn_wmma_f32_16x16x32_f16(false, am1, false, bm,
                                                    (short)0, c1, false, false);
    }

    // ---- D layout: VGPR r, lane<16 -> (M=r, N=lane); lane>=16 -> (M=8+r).
    #pragma unroll
    for (int r = 0; r < 8; ++r) {
        const int g0 = r + kh * 8;
        bcs[((size_t)(b * NB_ + g0))      * A_ + aCol] = c0[r];
        bcs[((size_t)(b * NB_ + 16 + g0)) * A_ + aCol] = c1[r];
    }
}

// =====================================================================
// K1: fused valid-mask / IoU / align.  One thread per (b,g,a).
// Reads bcs, overwrites it with align = bcs*vf * (iou*vf)^6.
// =====================================================================
__global__ __launch_bounds__(256)
void k1_align(const float*         __restrict__ pred_boxes,   // (B, A, 4)
              const float*         __restrict__ anchors,      // (A, 2)
              const float*         __restrict__ gt_boxes,     // (B, NB, 4)
              const unsigned char* __restrict__ gt_mask,      // (B, NB) bool
              float*               __restrict__ w_align,      // in: bcs -> out: align
              float*               __restrict__ w_iou,        // out: iou * vf
              unsigned char*       __restrict__ w_valid)      // out: vf
{
    const size_t total = (size_t)B_ * NB_ * A_;
    const size_t idx = (size_t)blockIdx.x * blockDim.x + threadIdx.x;
    if (idx >= total) return;
    const int a  = (int)(idx % A_);
    const int bg = (int)(idx / A_);
    const int g  = bg % NB_;
    const int b  = bg / NB_;

    // speculative prefetch of a later pred_boxes tile (global_prefetch_b8)
    __builtin_prefetch(pred_boxes + 4 * ((size_t)b * A_ + a) + 4096, 0, 1);

    const float2 ap = ((const float2*)anchors)[a];
    const float4 gb = ((const float4*)gt_boxes)[b * NB_ + g];
    const float4 pb = ((const float4*)pred_boxes)[(size_t)b * A_ + a];

    const float d = fminf(fminf(ap.x - gb.x, ap.y - gb.y),
                          fminf(gb.z - ap.x, gb.w - ap.y));
    const bool valid = (d > 1e-8f) && (gt_mask[b * NB_ + g] != 0);
    const float vf = valid ? 1.0f : 0.0f;

    float iw = fminf(gb.z, pb.z) - fmaxf(gb.x, pb.x); iw = fmaxf(iw, 0.0f);
    float ih = fminf(gb.w, pb.w) - fmaxf(gb.y, pb.y); ih = fmaxf(ih, 0.0f);
    const float inter = iw * ih;
    const float a1 = (gb.z - gb.x) * (gb.w - gb.y);
    const float a2 = (pb.z - pb.x) * (pb.w - pb.y);
    const float iou = fmaxf(inter / (a1 + a2 - inter + 1e-10f), 0.0f) * vf;

    const float bcs = w_align[idx] * vf;
    const float i2 = iou * iou;
    const float i4 = i2 * i2;

    w_align[idx] = bcs * (i4 * i2);   // bcs^1 * iou^6
    w_iou[idx]   = iou;
    w_valid[idx] = valid ? 1u : 0u;
}

// =====================================================================
// K2: per-(b,g) top-10 of align row -> mask.  The 8400-float row is staged
// into LDS by the Tensor Data Mover (one tensor_load_to_lds descriptor,
// issued by wave 0, waited with s_wait_tensorcnt), then 10 rounds of block
// argmax with first-index tie-break (selected indices are distinct, so this
// matches the reference's scatter-count dedup; vf zeroes invalid entries).
// =====================================================================
#define K2T 256
__global__ __launch_bounds__(K2T)
void k2_topk(const float*         __restrict__ w_align,
             const unsigned char* __restrict__ w_valid,
             float*               __restrict__ w_mask)
{
    __shared__ float s_val[A_];        // 33.6 KB (of 320 KB WGP LDS)
    __shared__ float s_red[K2T];
    __shared__ int   s_idx[K2T];

    const int bg = blockIdx.x;
    const int t  = threadIdx.x;
    const float*         row  = w_align + (size_t)bg * A_;
    const unsigned char* vrow = w_valid + (size_t)bg * A_;
    float*               mrow = w_mask  + (size_t)bg * A_;

#if HAVE_TDM
    if (t < 32) {   // wave 0 issues one TDM descriptor covering the whole row
        const unsigned long long gaddr = (unsigned long long)(uintptr_t)row;
        const unsigned int       laddr = (unsigned int)(uintptr_t)(void*)s_val;

        // ---- D# group 0: count=1 | lds_addr | global_addr[56:0] | type=2
        v4u g0 = {};
        g0[0] = 1u;                                   // count = 1 (valid)
        g0[1] = laddr;                                // lds_addr (bytes)
        g0[2] = (unsigned int)gaddr;                  // global_addr[31:0]
        g0[3] = (unsigned int)((gaddr >> 32) & 0x01FFFFFFull) | (2u << 30);

        // ---- D# group 1: data_size=4B, 1-row 2D tensor of A_ elements
        v8i g1 = {};
        g1[0] = (int)(2u << 16);                      // data_size = 2 (4 bytes)
        g1[1] = (int)(((unsigned)A_ & 0xFFFFu) << 16);// tensor_dim0[15:0] @bit48
        g1[2] = (int)(((unsigned)A_ >> 16) | (1u << 16)); // dim0 hi | tensor_dim1=1
        g1[3] = (int)(((unsigned)A_ & 0xFFFFu) << 16);// tile_dim0 = A_ @bit112
        g1[4] = 1;                                    // tile_dim1 = 1
        g1[5] = (int)A_;                              // tensor_dim0_stride[31:0]
        g1[6] = 0;
        g1[7] = 0;

        v4i g2 = {};                                  // 2D tensor: groups 2/3 unused
        v4i g3 = {};
#if __clang_major__ >= 23
        v8i g4 = {};
        __builtin_amdgcn_tensor_load_to_lds(g0, g1, g2, g3, g4, 0);
#else
        __builtin_amdgcn_tensor_load_to_lds(g0, g1, g2, g3, 0);
#endif
        __builtin_amdgcn_s_wait_tensorcnt(0);
    }
    for (int i = t; i < A_; i += K2T) mrow[i] = 0.0f;
    __syncthreads();
#else
    for (int i = t; i < A_; i += K2T) { s_val[i] = row[i]; mrow[i] = 0.0f; }
    __syncthreads();
#endif

    for (int it = 0; it < TOPK_; ++it) {
        float best = -1.0f;
        int   bi   = 0x7fffffff;
        for (int i = t; i < A_; i += K2T) {   // increasing i -> first-max wins
            const float v = s_val[i];
            if (v > best) { best = v; bi = i; }
        }
        s_red[t] = best; s_idx[t] = bi;
        __syncthreads();
        for (int off = K2T / 2; off > 0; off >>= 1) {
            if (t < off) {
                const float v2 = s_red[t + off];
                const int   i2 = s_idx[t + off];
                if (v2 > s_red[t] || (v2 == s_red[t] && i2 < s_idx[t])) {
                    s_red[t] = v2; s_idx[t] = i2;
                }
            }
            __syncthreads();
        }
        const int sel = s_idx[0];
        if (t == 0) {
            mrow[sel]  = vrow[sel] ? 1.0f : 0.0f;   // mask = topk * vf
            s_val[sel] = -3.0e38f;                  // remove from next rounds
        }
        __syncthreads();
    }
}

// =====================================================================
// K3: per-anchor conflict resolution + label/box/fg outputs.
// =====================================================================
__global__ __launch_bounds__(256)
void k3_assign(const float* __restrict__ w_iou,
               float*       __restrict__ w_mask,
               const int*   __restrict__ gt_labels,
               const float* __restrict__ gt_boxes,
               float*       __restrict__ out_labels,  // (B,A) as float
               float*       __restrict__ out_boxes,   // (B,A,4)
               float*       __restrict__ out_fg,      // (B,A) 0/1
               int*         __restrict__ w_label,
               float*       __restrict__ w_fgpos)
{
    const size_t idx = (size_t)blockIdx.x * blockDim.x + threadIdx.x;
    if (idx >= (size_t)B_ * A_) return;
    const int a = (int)(idx % A_);
    const int b = (int)(idx / A_);

    float m[NB_];
    float fg = 0.0f;
    #pragma unroll
    for (int g = 0; g < NB_; ++g) {
        m[g] = w_mask[((size_t)(b * NB_ + g)) * A_ + a];
        fg += m[g];
    }

    if (fg > 1.0f) {   // anchor claimed by >1 GT -> keep best-IoU GT (first max)
        float best = -1.0f; int bi = 0;
        #pragma unroll
        for (int g = 0; g < NB_; ++g) {
            const float v = w_iou[((size_t)(b * NB_ + g)) * A_ + a];
            if (v > best) { best = v; bi = g; }
        }
        #pragma unroll
        for (int g = 0; g < NB_; ++g) {
            const float nm = (g == bi) ? 1.0f : 0.0f;
            m[g] = nm;
            w_mask[((size_t)(b * NB_ + g)) * A_ + a] = nm;
        }
        fg = 1.0f;
    }

    // tgt = argmax over g of mask (first index of max; all-zero -> 0)
    int tgt = 0; float bm = m[0];
    #pragma unroll
    for (int g = 1; g < NB_; ++g)
        if (m[g] > bm) { bm = m[g]; tgt = g; }

    int lab = gt_labels[b * NB_ + tgt];
    lab = lab < 0 ? 0 : (lab > 80 ? 80 : lab);        // clip(0, NUM_CLASSES)
    const float4 tb = ((const float4*)gt_boxes)[b * NB_ + tgt];
    const float fgp = (fg > 0.0f) ? 1.0f : 0.0f;

    out_labels[idx]           = (float)lab;
    ((float4*)out_boxes)[idx] = tb;
    out_fg[idx]               = fgp;
    w_label[idx]              = lab;
    w_fgpos[idx]              = fgp;
}

// =====================================================================
// K4: pos_align/pos_iou per (b,g): max over A of align*mask, iou*mask.
// =====================================================================
__global__ __launch_bounds__(256)
void k4_pos(const float* __restrict__ w_align,
            const float* __restrict__ w_iou,
            const float* __restrict__ w_mask,
            float*       __restrict__ w_posalign,
            float*       __restrict__ w_posiou)
{
    __shared__ float sa[256], si[256];
    const int bg = blockIdx.x;
    const int t  = threadIdx.x;
    const float* ar = w_align + (size_t)bg * A_;
    const float* ir = w_iou   + (size_t)bg * A_;
    const float* mr = w_mask  + (size_t)bg * A_;
    float pa = 0.0f, pi = 0.0f;                  // all candidates >= 0
    for (int i = t; i < A_; i += 256) {
        const float mm = mr[i];
        pa = fmaxf(pa, ar[i] * mm);
        pi = fmaxf(pi, ir[i] * mm);
    }
    sa[t] = pa; si[t] = pi;
    __syncthreads();
    for (int off = 128; off > 0; off >>= 1) {
        if (t < off) {
            sa[t] = fmaxf(sa[t], sa[t + off]);
            si[t] = fmaxf(si[t], si[t + off]);
        }
        __syncthreads();
    }
    if (t == 0) { w_posalign[bg] = sa[0]; w_posiou[bg] = si[0]; }
}

// =====================================================================
// K5a: per-anchor norm = max_g align*mask*pos_iou/(pos_align+eps)
// =====================================================================
__global__ __launch_bounds__(256)
void k5a_norm(const float* __restrict__ w_align,
              const float* __restrict__ w_mask,
              const float* __restrict__ w_posalign,
              const float* __restrict__ w_posiou,
              float*       __restrict__ w_norm)
{
    const size_t idx = (size_t)blockIdx.x * blockDim.x + threadIdx.x;
    if (idx >= (size_t)B_ * A_) return;
    const int a = (int)(idx % A_);
    const int b = (int)(idx / A_);
    float nm = 0.0f;                             // candidates >= 0
    #pragma unroll
    for (int g = 0; g < NB_; ++g) {
        const int bg = b * NB_ + g;
        const size_t o = (size_t)bg * A_ + a;
        const float v = w_align[o] * w_mask[o];
        nm = fmaxf(nm, v * w_posiou[bg] / (w_posalign[bg] + 1e-8f));
    }
    w_norm[idx] = nm;
}

// =====================================================================
// K5b: target_scores = onehot(label) * (fg>0) * norm, coalesced over C.
// =====================================================================
__global__ __launch_bounds__(256)
void k5b_scores(const int*   __restrict__ w_label,
                const float* __restrict__ w_fgpos,
                const float* __restrict__ w_norm,
                float*       __restrict__ out_scores)   // (B, A, C)
{
    const size_t total = (size_t)B_ * A_ * C_;
    const size_t idx = (size_t)blockIdx.x * blockDim.x + threadIdx.x;
    if (idx >= total) return;
    const int    c  = (int)(idx % C_);
    const size_t ba = idx / C_;
    out_scores[idx] =
        (w_label[ba] == c && w_fgpos[ba] > 0.0f) ? w_norm[ba] : 0.0f;
}

// =====================================================================
// launch
// =====================================================================
extern "C" void kernel_launch(void* const* d_in, const int* in_sizes, int n_in,
                              void* d_out, int out_size, void* d_ws, size_t ws_size,
                              hipStream_t stream)
{
    const float*         pred_scores = (const float*)d_in[0];         // (B,A,C)
    const float*         pred_boxes  = (const float*)d_in[1];         // (B,A,4)
    const float*         anchors     = (const float*)d_in[2];         // (A,2)
    const int*           gt_labels   = (const int*)d_in[3];           // (B,NB,1)
    const float*         gt_boxes    = (const float*)d_in[4];         // (B,NB,4)
    const unsigned char* gt_mask     = (const unsigned char*)d_in[5]; // (B,NB,1) bool

    const size_t BA   = (size_t)B_ * A_;
    const size_t nBGA = (size_t)B_ * NB_ * A_;

    // workspace carve-up (~115 MB)
    float*         w_align    = (float*)d_ws;               // bcs -> align
    float*         w_iou      = w_align + nBGA;
    float*         w_mask     = w_iou + nBGA;
    unsigned char* w_valid    = (unsigned char*)(w_mask + nBGA);
    float*         w_posalign = (float*)(w_valid + nBGA);   // nBGA mult of 16
    float*         w_posiou   = w_posalign + (size_t)B_ * NB_;
    float*         w_norm     = w_posiou + (size_t)B_ * NB_;
    float*         w_fgpos    = w_norm + BA;
    int*           w_label    = (int*)(w_fgpos + BA);

    // output carve-up: [labels BA][boxes 4BA][scores 80BA][fg BA]
    float* out_labels = (float*)d_out;
    float* out_boxes  = out_labels + BA;
    float* out_scores = out_boxes + 4 * BA;
    float* out_fg     = out_scores + (size_t)C_ * BA;

    // K0: bcs via one-hot GEMM on WMMA
    k0_bcs_wmma<<<dim3(A_ / 16, B_), 32, 0, stream>>>(pred_scores, gt_labels, w_align);

    // K1: fused valid / IoU / align
    {
        const size_t total = nBGA;
        k1_align<<<(unsigned)((total + 255) / 256), 256, 0, stream>>>(
            pred_boxes, anchors, gt_boxes, gt_mask,
            w_align, w_iou, w_valid);
    }

    // K2: top-10 per (b,g), row staged via Tensor Data Mover
    k2_topk<<<B_ * NB_, K2T, 0, stream>>>(w_align, w_valid, w_mask);

    // K3: per-anchor resolve + labels/boxes/fg
    k3_assign<<<(unsigned)((BA + 255) / 256), 256, 0, stream>>>(
        w_iou, w_mask, gt_labels, gt_boxes,
        out_labels, out_boxes, out_fg, w_label, w_fgpos);

    // K4: pos_align / pos_iou
    k4_pos<<<B_ * NB_, 256, 0, stream>>>(w_align, w_iou, w_mask, w_posalign, w_posiou);

    // K5a: per-anchor norm
    k5a_norm<<<(unsigned)((BA + 255) / 256), 256, 0, stream>>>(
        w_align, w_mask, w_posalign, w_posiou, w_norm);

    // K5b: 80-wide scaled one-hot scores
    {
        const size_t total = BA * C_;
        k5b_scores<<<(unsigned)((total + 255) / 256), 256, 0, stream>>>(
            w_label, w_fgpos, w_norm, out_scores);
    }
}
